// GraphBatchNetAMP_83537113907556
// MI455X (gfx1250) — matrix operand order repaired and checked by reading, not compiled, not run on hardware
//
#include <hip/hip_runtime.h>
#include <hip/hip_bf16.h>

// ---------------------------------------------------------------------------
// GraphBatchNet on MI455X (gfx1250), wave32 + WMMA bf16 (f32 accumulate).
//
// Math reduction: H only feeds H.mean(), and agg = scatter(m,src)+scatter(m,dst)
// so sum(agg) = 2*sum_edges(m). edge_ctx is linear in E@pW, so
// edge_ctx = (sum gate*E)@pW + (sum gate)*pb. => no scatter needed, only sums.
//
// GEMM blocking: kt-outer / nt-inner over 4 independent accumulators
// (two sequential nt-half passes) -> back-to-back independent WMMAs without
// register spills.
// ---------------------------------------------------------------------------

#define NODE_DIM 128
#define HIDDEN   128
#define EDGE_DIM 16

// workspace layout (bytes)
#define ACC_STRIDE 512                      // floats per batch: [0,128) SHx, [128,256) Sm, [256,272) SgE, [272] Sg
#define O_ACC   0
#define O_NW1T  4096
#define O_NW2T  (O_NW1T + 128*128*2)
#define O_EW2T  (O_NW2T + 128*128*2)
#define O_EW1T  (O_EW2T + 128*128*2)
#define EW1T_K  288                         // 272 padded to 9*32

typedef __attribute__((ext_vector_type(16))) __bf16 v16bf;
typedef __attribute__((ext_vector_type(8)))  float  v8f;

__device__ __forceinline__ float relu0(float x) { return x > 0.f ? x : 0.f; }

// A fragment (16x32 bf16, M x K) from row-major LDS tile.
// lanes 0-15: rows 0-15, K 0..15; lanes 16-31: same rows, K 16..31
__device__ __forceinline__ v16bf load_A_lds(const __bf16* tile, int stride, int kbase, int lane) {
  int m = lane & 15, hi = lane >> 4;
  const __bf16* row = tile + m * stride + kbase;
  v16bf a;
#pragma unroll
  for (int j = 0; j < 8; ++j) {
    int k = hi * 16 + (j << 1);
    a[2 * j]     = row[k];
    a[2 * j + 1] = row[k + 1];
  }
  return a;
}

// B fragment (32x16 bf16, K x N) from pre-transposed weights Wt[n][K]:
// lane n<16 holds K=kbase..kbase+15 of column (nbase+n); lane n+16 holds K+16..+31.
// One contiguous, 32B-aligned 16-element load per lane.
__device__ __forceinline__ v16bf load_B_w(const __bf16* Wt, int Kstride, int kbase, int nbase, int lane) {
  int n = nbase + (lane & 15);
  int koff = (lane >> 4) << 4;
  return *(const v16bf*)(Wt + (size_t)n * Kstride + kbase + koff);
}

__device__ __forceinline__ v8f wmma_bf16(v16bf a, v16bf b, v8f c) {
  return __builtin_amdgcn_wmma_f32_16x16x32_bf16(false, a, false, b, (short)0, c, false, false);
}

// 4 independent WMMAs sharing one A fragment (columns (nt0..nt0+3)*16)
__device__ __forceinline__ void wmma4(v8f c[4], v16bf a, const __bf16* Wt, int Kstride,
                                      int kbase, int nt0, int lane) {
#pragma unroll
  for (int i = 0; i < 4; ++i)
    c[i] = wmma_bf16(a, load_B_w(Wt, Kstride, kbase, (nt0 + i) * 16, lane), c[i]);
}

// ---------------------------------------------------------------------------
// prep: transpose+convert weights to bf16, zero accumulators
// ---------------------------------------------------------------------------
__global__ void gbn_prep(const float* __restrict__ nW1, const float* __restrict__ nW2,
                         const float* __restrict__ eW1, const float* __restrict__ eW2,
                         char* __restrict__ ws, int nAccFloats) {
  int tid = blockIdx.x * blockDim.x + threadIdx.x;
  float* acc = (float*)(ws + O_ACC);
  if (tid < nAccFloats) acc[tid] = 0.f;
  __bf16* nW1t = (__bf16*)(ws + O_NW1T);
  __bf16* nW2t = (__bf16*)(ws + O_NW2T);
  __bf16* eW2t = (__bf16*)(ws + O_EW2T);
  __bf16* eW1t = (__bf16*)(ws + O_EW1T);
  if (tid < 128 * 128) {
    int n = tid >> 7, k = tid & 127;
    nW1t[n * 128 + k] = (__bf16)nW1[k * 128 + n];
    nW2t[n * 128 + k] = (__bf16)nW2[k * 128 + n];
    eW2t[n * 128 + k] = (__bf16)eW2[k * 128 + n];
  }
  if (tid < 128 * EW1T_K) {
    int n = tid / EW1T_K, k = tid % EW1T_K;
    eW1t[n * EW1T_K + k] = (k < 272) ? (__bf16)eW1[k * 128 + n] : (__bf16)0.f;
  }
}

// ---------------------------------------------------------------------------
// node kernel: Hx = relu(relu(X@nW1+b1)@nW2+b2); accumulate column sums.
// 128 threads = 4 waves; each wave owns a 16-node tile.
// ---------------------------------------------------------------------------
__global__ void gbn_node(const float* __restrict__ X,
                         const float* __restrict__ nb1, const float* __restrict__ nb2,
                         const char* __restrict__ ws, float* __restrict__ acc, int N) {
  __shared__ __bf16 Xt[64 * 128];
  __shared__ __bf16 H1[4][16 * 128];
  __shared__ float  sum128[128];

  const int tid = threadIdx.x, lane = tid & 31, wave = tid >> 5;
  const int b = blockIdx.y;
  const int nodeBase = blockIdx.x * 64;
  const __bf16* nW1t = (const __bf16*)(ws + O_NW1T);
  const __bf16* nW2t = (const __bf16*)(ws + O_NW2T);
  const float* Xb = X + (size_t)b * N * NODE_DIM;

  __builtin_prefetch(nW1t, 0, 2);
  __builtin_prefetch(nW2t, 0, 2);

  if (tid < 128) sum128[tid] = 0.f;
  for (int idx = tid; idx < 64 * 128; idx += 128) {
    int row = idx >> 7, col = idx & 127;
    int gr = nodeBase + row;
    Xt[idx] = (__bf16)((gr < N) ? Xb[(size_t)gr * 128 + col] : 0.f);
  }
  __syncthreads();

  const int nn = lane & 15, hi = lane >> 4;
  const int tileRow = nodeBase + wave * 16;

  // ---- layer 1: two nt-half passes of 4 column tiles ----
#pragma unroll 1
  for (int half = 0; half < 2; ++half) {
    const int nt0 = half * 4;
    v8f c[4];
#pragma unroll
    for (int i = 0; i < 4; ++i) {
      float bias = nb1[(nt0 + i) * 16 + nn];
#pragma unroll
      for (int r = 0; r < 8; ++r) c[i][r] = bias;
    }
#pragma unroll
    for (int kt = 0; kt < 4; ++kt) {
      v16bf a = load_A_lds(&Xt[wave * 16 * 128], 128, kt * 32, lane);
      wmma4(c, a, nW1t, 128, kt * 32, nt0, lane);
    }
#pragma unroll
    for (int i = 0; i < 4; ++i)
#pragma unroll
      for (int r = 0; r < 8; ++r)
        H1[wave][(r + hi * 8) * 128 + (nt0 + i) * 16 + nn] = (__bf16)relu0(c[i][r]);
  }

  // ---- layer 2 + masked column sum ----
#pragma unroll 1
  for (int half = 0; half < 2; ++half) {
    const int nt0 = half * 4;
    v8f c[4];
#pragma unroll
    for (int i = 0; i < 4; ++i) {
      float bias = nb2[(nt0 + i) * 16 + nn];
#pragma unroll
      for (int r = 0; r < 8; ++r) c[i][r] = bias;
    }
#pragma unroll
    for (int kt = 0; kt < 4; ++kt) {
      v16bf a = load_A_lds(&H1[wave][0], 128, kt * 32, lane);
      wmma4(c, a, nW2t, 128, kt * 32, nt0, lane);
    }
#pragma unroll
    for (int i = 0; i < 4; ++i) {
      float s = 0.f;
#pragma unroll
      for (int r = 0; r < 8; ++r)
        if (tileRow + r + hi * 8 < N) s += relu0(c[i][r]);
      atomicAdd(&sum128[(nt0 + i) * 16 + nn], s);
    }
  }
  __syncthreads();
  if (tid < 128) atomicAdd(&acc[b * ACC_STRIDE + tid], sum128[tid]);
}

// ---------------------------------------------------------------------------
// edge kernel: m = relu(relu([X[src],X[dst],E]@eW1+b1)@eW2+b2)*gate;
// accumulate sum(m), sum(gate*E), sum(gate). 128 threads = 4 waves x 16 edges.
// ---------------------------------------------------------------------------
__global__ void gbn_edge(const float* __restrict__ X, const float* __restrict__ E,
                         const int* __restrict__ edges,
                         const float* __restrict__ eb1, const float* __restrict__ eb2,
                         const char* __restrict__ ws, float* __restrict__ acc,
                         const float* __restrict__ gsp, int N, int M) {
  __shared__ __bf16 Xs[4][16 * 128];
  __shared__ __bf16 Xd[4][16 * 128];
  __shared__ __bf16 Ep[4][16 * 32];
  __shared__ float  Ef[4][16 * 16];
  __shared__ float  gateS[4][16];
  __shared__ int    srcS[4][16], dstS[4][16];
  __shared__ __bf16 H1[4][16 * 128];
  __shared__ float  sum128[128];
  __shared__ float  sumE[16];
  __shared__ float  sumG;

  const int tid = threadIdx.x, lane = tid & 31, wave = tid >> 5;
  const int b = blockIdx.y;
  const int base = (blockIdx.x * 4 + wave) * 16;
  int nvalid = M - base;
  nvalid = nvalid < 0 ? 0 : (nvalid > 16 ? 16 : nvalid);

  const __bf16* eW1t = (const __bf16*)(ws + O_EW1T);
  const __bf16* eW2t = (const __bf16*)(ws + O_EW2T);
  const float* Xb = X + (size_t)b * N * NODE_DIM;
  const float* Eb = E + (size_t)b * M * EDGE_DIM;
  const int*   edb = edges + (size_t)b * M * 2;
  const float gs = *gsp;

  __builtin_prefetch(eW1t, 0, 2);
  __builtin_prefetch(eW2t, 0, 2);

  if (tid < 128) sum128[tid] = 0.f;
  if (tid < 16) sumE[tid] = 0.f;
  if (tid == 16) sumG = 0.f;

  // edge indices (clamped for tail)
  if (lane < 16) {
    int ge = base + lane; ge = ge < M - 1 ? ge : M - 1;
    srcS[wave][lane] = edb[ge * 2 + 0];
    dstS[wave][lane] = edb[ge * 2 + 1];
  }
  // E tile: fp32 copy + bf16 zero-padded (16 -> 32 cols)
  for (int idx = lane; idx < 16 * 16; idx += 32) {
    int e = idx >> 4, col = idx & 15;
    int ge = base + e; ge = ge < M - 1 ? ge : M - 1;
    float v = Eb[(size_t)ge * EDGE_DIM + col];
    Ef[wave][idx] = v;
    Ep[wave][e * 32 + col] = (__bf16)v;
    Ep[wave][e * 32 + 16 + col] = (__bf16)0.f;
  }
  if (lane < 16) {
    float g = 1.f + gs * Ef[wave][lane * 16 + 2];
    gateS[wave][lane] = fminf(fmaxf(g, 0.f), 3.f);
  }
  // gather X[src], X[dst] -> bf16 LDS
  for (int idx = lane; idx < 16 * 128; idx += 32) {
    int e = idx >> 7, col = idx & 127;
    Xs[wave][idx] = (__bf16)Xb[(size_t)srcS[wave][e] * 128 + col];
    Xd[wave][idx] = (__bf16)Xb[(size_t)dstS[wave][e] * 128 + col];
  }
  __syncthreads();

  const int nn = lane & 15, hi = lane >> 4;

  // ---- layer 1: K = 288 (128 src + 128 dst + 16 E padded to 32) ----
#pragma unroll 1
  for (int half = 0; half < 2; ++half) {
    const int nt0 = half * 4;
    v8f c[4];
#pragma unroll
    for (int i = 0; i < 4; ++i) {
      float bias = eb1[(nt0 + i) * 16 + nn];
#pragma unroll
      for (int r = 0; r < 8; ++r) c[i][r] = bias;
    }
#pragma unroll
    for (int kt = 0; kt < 9; ++kt) {
      v16bf a;
      if (kt < 4)      a = load_A_lds(&Xs[wave][0], 128, kt * 32, lane);
      else if (kt < 8) a = load_A_lds(&Xd[wave][0], 128, (kt - 4) * 32, lane);
      else             a = load_A_lds(&Ep[wave][0], 32, 0, lane);
      wmma4(c, a, eW1t, EW1T_K, kt * 32, nt0, lane);
    }
#pragma unroll
    for (int i = 0; i < 4; ++i)
#pragma unroll
      for (int r = 0; r < 8; ++r)
        H1[wave][(r + hi * 8) * 128 + (nt0 + i) * 16 + nn] = (__bf16)relu0(c[i][r]);
  }

  // ---- layer 2 + gate + masked column sum ----
#pragma unroll 1
  for (int half = 0; half < 2; ++half) {
    const int nt0 = half * 4;
    v8f c[4];
#pragma unroll
    for (int i = 0; i < 4; ++i) {
      float bias = eb2[(nt0 + i) * 16 + nn];
#pragma unroll
      for (int r = 0; r < 8; ++r) c[i][r] = bias;
    }
#pragma unroll
    for (int kt = 0; kt < 4; ++kt) {
      v16bf a = load_A_lds(&H1[wave][0], 128, kt * 32, lane);
      wmma4(c, a, eW2t, 128, kt * 32, nt0, lane);
    }
#pragma unroll
    for (int i = 0; i < 4; ++i) {
      float s = 0.f;
#pragma unroll
      for (int r = 0; r < 8; ++r) {
        int e = r + hi * 8;
        float v = relu0(c[i][r]) * gateS[wave][e];
        if (e < nvalid) s += v;
      }
      atomicAdd(&sum128[(nt0 + i) * 16 + nn], s);
    }
  }

  // gate*E and gate reductions
  if (lane < 16) {
    float se = 0.f;
    for (int e = 0; e < nvalid; ++e) se += gateS[wave][e] * Ef[wave][e * 16 + lane];
    atomicAdd(&sumE[lane], se);
  } else if (lane == 16) {
    float sg = 0.f;
    for (int e = 0; e < nvalid; ++e) sg += gateS[wave][e];
    atomicAdd(&sumG, sg);
  }
  __syncthreads();
  if (tid < 128) atomicAdd(&acc[b * ACC_STRIDE + 128 + tid], sum128[tid]);
  if (tid < 16)  atomicAdd(&acc[b * ACC_STRIDE + 256 + tid], sumE[tid]);
  if (tid == 16) atomicAdd(&acc[b * ACC_STRIDE + 272], sumG);
}

// ---------------------------------------------------------------------------
// final head: feat=[ (SHx+2*Sm)/N , (SgE@pW + Sg*pb)/(M+1e-6) ] -> 64 -> 2
// ---------------------------------------------------------------------------
__global__ void gbn_final(const float* __restrict__ acc,
                          const float* __restrict__ pW, const float* __restrict__ pb,
                          const float* __restrict__ rW1, const float* __restrict__ rb1,
                          const float* __restrict__ rW2, const float* __restrict__ rb2,
                          float* __restrict__ out, int N, int M) {
  __shared__ float feat[256];
  __shared__ float r1[64];
  const int b = blockIdx.x, tid = threadIdx.x;
  const float* a = acc + b * ACC_STRIDE;
  if (tid < 128) {
    feat[tid] = (a[tid] + 2.f * a[128 + tid]) / (float)N;
  } else if (tid < 256) {
    int n = tid - 128;
    float s = a[272] * pb[n];
    for (int k = 0; k < 16; ++k) s += a[256 + k] * pW[k * 128 + n];
    feat[tid] = s / ((float)M + 1e-6f);
  }
  __syncthreads();
  if (tid < 64) {
    float s = rb1[tid];
    for (int k = 0; k < 256; ++k) s += feat[k] * rW1[k * 64 + tid];
    r1[tid] = fmaxf(s, 0.f);
  }
  __syncthreads();
  if (tid < 2) {
    float s = rb2[tid];
    for (int k = 0; k < 64; ++k) s += r1[k] * rW2[k * 2 + tid];
    out[b * 2 + tid] = s;
  }
}

// ---------------------------------------------------------------------------
extern "C" void kernel_launch(void* const* d_in, const int* in_sizes, int n_in,
                              void* d_out, int out_size, void* d_ws, size_t ws_size,
                              hipStream_t stream) {
  const float* X     = (const float*)d_in[0];
  const float* E     = (const float*)d_in[1];
  const int*   edges = (const int*)d_in[2];
  const float* nW1   = (const float*)d_in[3];
  const float* nb1   = (const float*)d_in[4];
  const float* nW2   = (const float*)d_in[5];
  const float* nb2   = (const float*)d_in[6];
  const float* eW1   = (const float*)d_in[7];
  const float* eb1   = (const float*)d_in[8];
  const float* eW2   = (const float*)d_in[9];
  const float* eb2   = (const float*)d_in[10];
  const float* pW    = (const float*)d_in[11];
  const float* pb    = (const float*)d_in[12];
  const float* rW1   = (const float*)d_in[13];
  const float* rb1   = (const float*)d_in[14];
  const float* rW2   = (const float*)d_in[15];
  const float* rb2   = (const float*)d_in[16];
  const float* gsp   = (const float*)d_in[17];

  const int B = out_size / 2;
  const int N = in_sizes[0] / (B * NODE_DIM);
  const int M = in_sizes[1] / (B * EDGE_DIM);

  char*  ws  = (char*)d_ws;
  float* acc = (float*)(ws + O_ACC);

  int prepElems = 128 * EW1T_K;  // largest index space
  gbn_prep<<<(prepElems + 255) / 256, 256, 0, stream>>>(nW1, nW2, eW1, eW2, ws, 2 * ACC_STRIDE);

  dim3 gN((N + 63) / 64, B);
  gbn_node<<<gN, 128, 0, stream>>>(X, nb1, nb2, ws, acc, N);

  dim3 gE((M + 63) / 64, B);
  gbn_edge<<<gE, 128, 0, stream>>>(X, E, edges, eb1, eb2, ws, acc, gsp, N, M);

  gbn_final<<<B, 256, 0, stream>>>(acc, pW, pb, rW1, rb1, rW2, rb2, (float*)d_out, N, M);
}
